// ScaledDotProductAttention_17128329577095
// MI455X (gfx1250) — matrix operand order repaired and checked by reading, not compiled, not run on hardware
//
#include <hip/hip_runtime.h>
#include <hip/hip_fp16.h>

typedef __attribute__((ext_vector_type(16))) _Float16 v16h;
typedef __attribute__((ext_vector_type(8)))  _Float16 h8;
typedef __attribute__((ext_vector_type(4)))  _Float16 h4;
typedef __attribute__((ext_vector_type(8)))  float    v8f;
typedef __attribute__((ext_vector_type(4)))  float    f4;
typedef unsigned char uc8 __attribute__((ext_vector_type(8)));

#define S_LEN 2048
#define DK    64
#define BH_N  16
#define KT    64           // keys per iteration
#define QW    16           // q rows per wave
#define NWAVE 8
#define QB    (QW * NWAVE) // 128 q rows per block

#define KSTR  72           // Klds row stride (halves) -> 144B rows, 16B aligned
#define VSTR  72           // Vt  row stride (halves)
#define PSTR  72           // Plds row stride (halves)

// 1/sqrt(64) * log2(e): scores land directly in log2 domain
#define SCALE_LOG2E 0.18033688011112042f

// raw v_exp_f32: args are <= 0; sub-denormal results flush to 0, which is
// exactly the flash-attention-desired behavior (avoids libm's cndmask+ldexp fixup)
__device__ __forceinline__ float exp2_raw(float x) {
    return __builtin_amdgcn_exp2f(x);
}

__global__ __launch_bounds__(256) void fa_wmma_kernel(
    const float* __restrict__ q, const float* __restrict__ k,
    const float* __restrict__ v, const unsigned char* __restrict__ mask,
    float* __restrict__ out)
{
    __shared__ _Float16 Klds[KT * KSTR];              // K[key][d]   f16
    __shared__ _Float16 Vt[DK * VSTR];                // V^T[d][key] f16
    __shared__ _Float16 Plds[NWAVE * QW * PSTR];      // P[q][key]   f16, per wave

    const int tid  = threadIdx.x;
    const int lane = tid & 31;
    const int wave = tid >> 5;
    const int n    = lane & 15;   // q-row (N dim) / A-row selector
    const int hi   = lane >> 4;   // half-wave selector

    const int bh   = blockIdx.x / (S_LEN / QB);
    const int qblk = blockIdx.x % (S_LEN / QB);
    const int q0   = qblk * QB + wave * QW;

    const size_t base = (size_t)bh * S_LEN * DK;

    // ---- Q as B-fragments for S^T = K * Q^T (pre-scaled, registers, loaded once)
    v16h bq[2];
    {
        const float* qp = q + base + (size_t)(q0 + n) * DK;
        #pragma unroll
        for (int s = 0; s < 2; ++s) {
            const int off = 32 * s + 16 * hi;
            #pragma unroll
            for (int i = 0; i < 16; ++i)
                bq[s][i] = (_Float16)(qp[off + i] * SCALE_LOG2E);
        }
    }

    v8f o[4];
    #pragma unroll
    for (int t = 0; t < 4; ++t) o[t] = (v8f){};
    float mval = -1e30f, lval = 0.0f;

    const unsigned char* mrowp =
        mask + ((size_t)bh * S_LEN + (size_t)(q0 + n)) * S_LEN + 8 * hi;

    // staging coordinates
    const int krow = tid >> 2;            // 0..63
    const int kcol = (tid & 3) * 16;      // 0..48
    const int vqd  = tid >> 4;            // 0..15 -> rows 4*vqd..+3
    const int vcol = (tid & 15) * 4;      // 0..60

    for (int k0 = 0; k0 < S_LEN; k0 += KT) {
        // ---- cooperative stage: K tile f16 row-major, V tile f16 transposed ----
        {
            const float* kp = k + base + (size_t)(k0 + krow) * DK + kcol;
            const float* vp = v + base + (size_t)(k0 + 4 * vqd) * DK + vcol;
            if (k0 + KT < S_LEN) {
                __builtin_prefetch(kp + KT * DK, 0, 0);
                __builtin_prefetch(vp + KT * DK, 0, 0);
            }
            h8 kh0, kh1;
            #pragma unroll
            for (int i = 0; i < 8; ++i) {
                kh0[i] = (_Float16)kp[i];
                kh1[i] = (_Float16)kp[8 + i];
            }
            *(h8*)&Klds[krow * KSTR + kcol]     = kh0;
            *(h8*)&Klds[krow * KSTR + kcol + 8] = kh1;
            #pragma unroll
            for (int i = 0; i < 4; ++i) {       // column d = vcol+i
                h4 tv;
                #pragma unroll
                for (int rr = 0; rr < 4; ++rr)
                    tv[rr] = (_Float16)vp[rr * DK + i];
                *(h4*)&Vt[(vcol + i) * VSTR + 4 * vqd] = tv;
            }
        }
        __syncthreads();

        // ---- S^T = K(64x64) * Q^T(64x16): 4 key M-tiles x 2 K-steps ----
        v8f st[4];
        #pragma unroll
        for (int t = 0; t < 4; ++t) st[t] = (v8f){};
        #pragma unroll
        for (int s = 0; s < 2; ++s) {
            #pragma unroll
            for (int t = 0; t < 4; ++t) {
                const _Float16* ap = &Klds[(16 * t + n) * KSTR + 32 * s + 16 * hi];
                h8 x = *(const h8*)ap, y = *(const h8*)(ap + 8);
                v16h a;
                #pragma unroll
                for (int i = 0; i < 8; ++i) { a[i] = x[i]; a[8 + i] = y[i]; }
                st[t] = __builtin_amdgcn_wmma_f32_16x16x32_f16(
                            false, a, false, bq[s], (short)0, st[t], false, false);
            }
        }

        // ---- mask (branch around the all-zero common case) ----
        // element r of tile t is key k0 + 16t + 8hi + r
        union { unsigned long long u; uc8 b; } mm[4];
        #pragma unroll
        for (int t = 0; t < 4; ++t)
            mm[t].u = *(const unsigned long long*)(mrowp + k0 + 16 * t);
        if (__any((mm[0].u | mm[1].u | mm[2].u | mm[3].u) != 0ULL)) {
            #pragma unroll
            for (int t = 0; t < 4; ++t)
                #pragma unroll
                for (int r = 0; r < 8; ++r)
                    st[t][r] = mm[t].b[r] ? -1e30f : st[t][r];
        }

        // ---- online softmax: lane owns q-row n (keys split with lane^16) ----
        float rmax = -1e30f;
        #pragma unroll
        for (int t = 0; t < 4; ++t)
            #pragma unroll
            for (int r = 0; r < 8; ++r)
                rmax = fmaxf(rmax, st[t][r]);
        rmax = fmaxf(rmax, __shfl_xor(rmax, 16, 32));
        const float mnew  = fmaxf(mval, rmax);
        const float alpha = exp2_raw(mval - mnew);

        float rs[4] = {0.0f, 0.0f, 0.0f, 0.0f};   // per-tile partial sums
        h8 ph[4];
        #pragma unroll
        for (int t = 0; t < 4; ++t) {
            #pragma unroll
            for (int r = 0; r < 8; ++r) {
                const float p = exp2_raw(st[t][r] - mnew);
                rs[t] += p;
                ph[t][r] = (_Float16)p;
            }
        }
        float rsum = (rs[0] + rs[1]) + (rs[2] + rs[3]);
        rsum += __shfl_xor(rsum, 16, 32);
        lval = lval * alpha + rsum;
        mval = mnew;
        #pragma unroll
        for (int t = 0; t < 4; ++t) o[t] = o[t] * alpha;

        // ---- store P[q][key]: row n, 8 consecutive halves per key tile ----
        {
            _Float16* pr = &Plds[(wave * QW + n) * PSTR];
            #pragma unroll
            for (int t = 0; t < 4; ++t)
                *(h8*)(pr + 16 * t + 8 * hi) = ph[t];
        }

        // ---- B-fragments: P^T chunks (lane n: col n, K = 32c+16hi..+16) ----
        v16h bp[2];
        #pragma unroll
        for (int c = 0; c < 2; ++c) {
            const _Float16* pp = &Plds[(wave * QW + n) * PSTR + 32 * c + 16 * hi];
            h8 x = *(const h8*)pp, y = *(const h8*)(pp + 8);
            #pragma unroll
            for (int i = 0; i < 8; ++i) { bp[c][i] = x[i]; bp[c][8 + i] = y[i]; }
        }

        // ---- O^T += V^T(64x64) * P^T(64x16): 4 d M-tiles x 2 K-chunks ----
        #pragma unroll
        for (int c = 0; c < 2; ++c) {
            #pragma unroll
            for (int t = 0; t < 4; ++t) {
                const _Float16* vp = &Vt[(16 * t + n) * VSTR + 32 * c + 16 * hi];
                h8 x = *(const h8*)vp, y = *(const h8*)(vp + 8);
                v16h av;
                #pragma unroll
                for (int i = 0; i < 8; ++i) { av[i] = x[i]; av[8 + i] = y[i]; }
                o[t] = __builtin_amdgcn_wmma_f32_16x16x32_f16(
                           false, av, false, bp[c], (short)0, o[t], false, false);
            }
        }
        __syncthreads();   // before next tile overwrites K/V staging
    }

    // ---- normalize and write out: lane n owns q-row n, d = 16t + 8hi + r ----
    const float inv = __builtin_amdgcn_rcpf(lval);
    float* op = out + base + (size_t)(q0 + n) * DK + 8 * hi;
    #pragma unroll
    for (int t = 0; t < 4; ++t) {
        f4 lo  = { o[t][0] * inv, o[t][1] * inv, o[t][2] * inv, o[t][3] * inv };
        f4 hi4 = { o[t][4] * inv, o[t][5] * inv, o[t][6] * inv, o[t][7] * inv };
        *(f4*)(op + 16 * t)     = lo;
        *(f4*)(op + 16 * t + 4) = hi4;
    }
}

extern "C" void kernel_launch(void* const* d_in, const int* in_sizes, int n_in,
                              void* d_out, int out_size, void* d_ws, size_t ws_size,
                              hipStream_t stream) {
    const float* q = (const float*)d_in[0];
    const float* k = (const float*)d_in[1];
    const float* v = (const float*)d_in[2];
    const unsigned char* mask = (const unsigned char*)d_in[3];
    float* out = (float*)d_out;

    const int blocks = BH_N * (S_LEN / QB);   // 256
    fa_wmma_kernel<<<blocks, 256, 0, stream>>>(q, k, v, mask, out);
}
